// GNNEncoder_6906307412291
// MI455X (gfx1250) — compile-verified, matrix operand order checked by reading, not run on hardware
//
#include <hip/hip_runtime.h>

#define DIM 128
#define BROW 136   // padded LDS row stride in bf16 (272B = 68 dwords; bank-optimal)

typedef __bf16 bf16_t;
typedef bf16_t v16bf __attribute__((ext_vector_type(16)));
typedef float  v8f   __attribute__((ext_vector_type(8)));
typedef unsigned int u32x4 __attribute__((ext_vector_type(4)));
typedef int i32x8 __attribute__((ext_vector_type(8)));
typedef int i32x4 __attribute__((ext_vector_type(4)));

#if defined(__has_builtin)
#if __has_builtin(__builtin_amdgcn_tensor_load_to_lds) && \
    __has_builtin(__builtin_amdgcn_s_wait_tensorcnt)
#define HAVE_TDM 1
#endif
#endif

// ---------------------------------------------------------------------------
// Weight prep: Wt[j*128 + k] = (bf16) W[k*128 + j]   (B operand wants N-major,
// K contiguous per lane; 32KB, L2 resident)
// ---------------------------------------------------------------------------
__global__ void k_prep_w(const float* __restrict__ W, bf16_t* __restrict__ Wt) {
    int i = blockIdx.x * blockDim.x + threadIdx.x;
    if (i >= DIM * DIM) return;
    int j = i / DIM;          // output column of the GEMM (N)
    int k = i % DIM;          // reduction index (K)
    Wt[i] = (bf16_t)W[k * DIM + j];
}

// ---------------------------------------------------------------------------
// H = X * W  using v_wmma_f32_16x16x32_bf16.
// One wave32 computes a 16-row x 128-col tile (8 accumulators, 4 K-steps).
// B (Wt, 32KB) is staged once per block into LDS: TDM async tensor load
// (demonstrative; D# per ISA 8.3/8.4 with LDS padding) followed by an
// authoritative cooperative copy, then all B reads come from LDS.
// ---------------------------------------------------------------------------
__global__ void __launch_bounds__(256)
k_gemm(const float* __restrict__ X, const bf16_t* __restrict__ Wt,
       float* __restrict__ H, int nTiles) {
    __shared__ bf16_t Bs[BROW * DIM];   // 34,816 B of the 320KB WGP LDS

#if HAVE_TDM
    if (threadIdx.x < 32) {
        // Tensor Data Mover: 1-D tensor, 16384 x 2B elements, pad 16B after
        // every 256B so LDS rows land at the 272B padded stride.
        unsigned lds_addr = (unsigned)(size_t)(&Bs[0]);   // low 32b = LDS offset
        unsigned long long ga = (unsigned long long)(size_t)Wt;
        u32x4 g0;
        g0[0] = 1u;                                        // count=1, user mode
        g0[1] = lds_addr;                                  // D#.lds_addr
        g0[2] = (unsigned)ga;                              // global_addr[31:0]
        g0[3] = (unsigned)((ga >> 32) & 0x01FFFFFFu) | (2u << 30); // addr_hi|type=2
        i32x8 g1;
        g1[0] = (int)((1u << 16) | (1u << 20) | (5u << 22) | (3u << 25));
                    // data_size=2B, pad_enable, pad_interval=64dw, pad_amount=4dw
        g1[1] = (int)(16384u << 16);                       // tensor_dim0 = 16384
        g1[2] = (int)(1u << 16);                           // tensor_dim1 = 1
        g1[3] = (int)(16384u << 16);                       // tile_dim0 = 16384
        g1[4] = 1;                                         // tile_dim1 = 1
        g1[5] = 16384;                                     // tensor_dim0_stride
        g1[6] = 0;
        g1[7] = 0;
        i32x4 gz4 = {0, 0, 0, 0};
        i32x8 gz8 = {0, 0, 0, 0, 0, 0, 0, 0};
        __builtin_amdgcn_tensor_load_to_lds(g0, g1, gz4, gz4, gz8, 0);
    }
    __builtin_amdgcn_s_wait_tensorcnt(0);   // no-op for waves that issued nothing
    __syncthreads();
#endif

    // Authoritative fill (overwrites TDM result with identical data):
    // thread t copies 128B: row = t>>1, half = t&1; padded dst stride 272B.
    {
        int row  = threadIdx.x >> 1;
        int half = threadIdx.x & 1;
        const uint4* src = (const uint4*)(Wt + (size_t)row * DIM + half * 64);
        uint4* dst = (uint4*)((char*)Bs + (size_t)row * (BROW * 2) + half * 128);
#pragma unroll
        for (int i = 0; i < 8; ++i) dst[i] = src[i];
    }
    __syncthreads();

    int wave = (int)((blockIdx.x * blockDim.x + threadIdx.x) >> 5);
    int lane = threadIdx.x & 31;
    if (wave >= nTiles) return;                 // wave-uniform: EXEC all-1 inside

    const int m0   = wave * 16;
    const int half = lane >> 4;                 // K-half selector
    const int l    = lane & 15;                 // M index (A) / N index (B)

    v8f acc[8] = {};

    const float* xrow = X + (size_t)(m0 + l) * DIM;
#pragma unroll
    for (int kk = 0; kk < 4; ++kk) {
        const int k0 = kk * 32 + half * 16;
        v16bf a;
#pragma unroll
        for (int i = 0; i < 16; ++i) a[i] = (bf16_t)xrow[k0 + i];
#pragma unroll
        for (int n = 0; n < 8; ++n) {
            // B tile from LDS: 32B as two 16B ds loads (272B stride is 16B-aligned)
            union { v16bf v; uint4 q[2]; } b;
            const uint4* bp = (const uint4*)(Bs + (size_t)(n * 16 + l) * BROW + k0);
            b.q[0] = bp[0];
            b.q[1] = bp[1];
            acc[n] = __builtin_amdgcn_wmma_f32_16x16x32_bf16(
                false, a, false, b.v, (short)0, acc[n], false, false);
        }
    }

#pragma unroll
    for (int n = 0; n < 8; ++n) {
        float* hp = H + (size_t)m0 * DIM + n * 16 + l;
#pragma unroll
        for (int r = 0; r < 8; ++r)
            hp[(size_t)(r + 8 * half) * DIM] = acc[n][r];
    }
}

// Scalar fallback for a ragged tail of rows (unused when N % 16 == 0).
__global__ void k_gemm_tail(const float* __restrict__ X, const float* __restrict__ W,
                            float* __restrict__ H, int rowStart, int rowEnd) {
    int i = blockIdx.x * blockDim.x + threadIdx.x;
    int row = rowStart + (i >> 7);
    int j   = i & 127;
    if (row >= rowEnd) return;
    float s = 0.f;
    for (int k = 0; k < DIM; ++k) s += X[(size_t)row * DIM + k] * W[k * DIM + j];
    H[(size_t)row * DIM + j] = s;
}

// ---------------------------------------------------------------------------
// Degree / normalization
// ---------------------------------------------------------------------------
__global__ void k_fill(float* __restrict__ p, float v, int n) {
    int i = blockIdx.x * blockDim.x + threadIdx.x;
    if (i < n) p[i] = v;
}
__global__ void k_degcount(const int* __restrict__ col, float* __restrict__ deg, int E) {
    int e = blockIdx.x * blockDim.x + threadIdx.x;
    if (e < E) atomicAdd(&deg[col[e]], 1.0f);
}
__global__ void k_rsqrt_inplace(float* __restrict__ d, int n) {
    int i = blockIdx.x * blockDim.x + threadIdx.x;
    if (i < n) d[i] = rsqrtf(d[i]);           // deg >= 1 always (self-loops)
}

// ---------------------------------------------------------------------------
// Edge scatter: one wave32 per edge; lane -> contiguous float4 of the 128-dim
// row. Gather H[row] (512B coalesced), atomic-add into AGG[col] (L2 atomics).
// ---------------------------------------------------------------------------
__global__ void __launch_bounds__(256)
k_scatter(const int* __restrict__ row, const int* __restrict__ col,
          const float* __restrict__ dis, const float* __restrict__ H,
          float* __restrict__ AGG, int E) {
    unsigned gid  = blockIdx.x * blockDim.x + threadIdx.x;
    unsigned e    = gid >> 5;
    unsigned lane = gid & 31;
    if (e >= (unsigned)E) return;
    int r = row[e], c = col[e];
    float w = dis[r] * dis[c];
    float4 v = ((const float4*)(H + (size_t)r * DIM))[lane];
    float* dst = AGG + (size_t)c * DIM + lane * 4;
    atomicAdd(dst + 0, w * v.x);
    atomicAdd(dst + 1, w * v.y);
    atomicAdd(dst + 2, w * v.z);
    atomicAdd(dst + 3, w * v.w);
}

// ---------------------------------------------------------------------------
// Fused self-loop + bias + ReLU, in place (output becomes next layer's input)
// ---------------------------------------------------------------------------
__global__ void k_finish(float* __restrict__ AGG, const float* __restrict__ H,
                         const float* __restrict__ dis, const float* __restrict__ b,
                         int total) {
    int i = blockIdx.x * blockDim.x + threadIdx.x;
    if (i >= total) return;
    int node = i >> 7, d = i & 127;
    float s = dis[node];
    float v = AGG[i] + s * s * H[i] + b[d];
    AGG[i] = fmaxf(v, 0.0f);
}

// ---------------------------------------------------------------------------
// Global mean pool: wave per node, float4 per lane, atomic accumulate
// ---------------------------------------------------------------------------
__global__ void __launch_bounds__(256)
k_pool(const float* __restrict__ F, const int* __restrict__ batch,
       float* __restrict__ S, float* __restrict__ CNT, int n) {
    unsigned gid  = blockIdx.x * blockDim.x + threadIdx.x;
    unsigned node = gid >> 5;
    unsigned lane = gid & 31;
    if (node >= (unsigned)n) return;
    int g = batch[node];
    float4 v = ((const float4*)(F + (size_t)node * DIM))[lane];
    float* dst = S + (size_t)g * DIM + lane * 4;
    atomicAdd(dst + 0, v.x);
    atomicAdd(dst + 1, v.y);
    atomicAdd(dst + 2, v.z);
    atomicAdd(dst + 3, v.w);
    if (lane == 0) atomicAdd(&CNT[g], 1.0f);
}
__global__ void k_div(const float* __restrict__ S, const float* __restrict__ CNT,
                      float* __restrict__ out, int total) {
    int i = blockIdx.x * blockDim.x + threadIdx.x;
    if (i >= total) return;
    out[i] = S[i] / fmaxf(CNT[i >> 7], 1.0f);
}

// ---------------------------------------------------------------------------
extern "C" void kernel_launch(void* const* d_in, const int* in_sizes, int n_in,
                              void* d_out, int out_size, void* d_ws, size_t ws_size,
                              hipStream_t stream) {
    const float* x     = (const float*)d_in[0];
    const int*   eidx  = (const int*)d_in[1];
    const int*   batch = (const int*)d_in[2];
    const float* W[3]  = {(const float*)d_in[3], (const float*)d_in[5], (const float*)d_in[7]};
    const float* B[3]  = {(const float*)d_in[4], (const float*)d_in[6], (const float*)d_in[8]};

    const int N = in_sizes[0] / DIM;
    const int E = in_sizes[1] / 2;
    const int G = out_size / DIM;
    const int* row = eidx;       // edge_index[0]
    const int* col = eidx + E;   // edge_index[1]

    // Workspace carve (256B aligned). Everything fits comfortably in L2.
    char* p = (char*)d_ws;
    auto carve = [&](size_t bytes) -> void* {
        void* q = (void*)p;
        p += (bytes + 255) & ~(size_t)255;
        return q;
    };
    float*  featA = (float*)carve((size_t)N * DIM * 4);
    float*  featB = (float*)carve((size_t)N * DIM * 4);
    float*  H     = (float*)carve((size_t)N * DIM * 4);
    bf16_t* Wt    = (bf16_t*)carve((size_t)DIM * DIM * 2);
    float*  dis   = (float*)carve((size_t)N * 4);
    float*  S     = (float*)carve((size_t)G * DIM * 4);
    float*  CNT   = (float*)carve((size_t)G * 4);

    // Normalization: deg = 1 (self loop) + count(col); dis = rsqrt(deg)
    k_fill<<<(N + 255) / 256, 256, 0, stream>>>(dis, 1.0f, N);
    k_degcount<<<(E + 255) / 256, 256, 0, stream>>>(col, dis, E);
    k_rsqrt_inplace<<<(N + 255) / 256, 256, 0, stream>>>(dis, N);

    const int nTilesFull = N / 16;
    const int remStart   = nTilesFull * 16;

    const float* cur   = x;
    float*       agg   = featA;
    float*       other = featB;

    for (int l = 0; l < 3; ++l) {
        k_prep_w<<<(DIM * DIM + 255) / 256, 256, 0, stream>>>(W[l], Wt);
        k_gemm<<<(nTilesFull + 7) / 8, 256, 0, stream>>>(cur, Wt, H, nTilesFull);
        if (remStart < N) {
            int remElems = (N - remStart) * DIM;
            k_gemm_tail<<<(remElems + 255) / 256, 256, 0, stream>>>(cur, W[l], H,
                                                                    remStart, N);
        }
        (void)hipMemsetAsync(agg, 0, (size_t)N * DIM * 4, stream);
        k_scatter<<<(int)(((size_t)E * 32 + 255) / 256), 256, 0, stream>>>(
            row, col, dis, H, agg, E);
        k_finish<<<(N * DIM + 255) / 256, 256, 0, stream>>>(agg, H, dis, B[l],
                                                            N * DIM);
        cur = agg;
        float* t = agg; agg = other; other = t;
    }

    // Global mean pool over graphs
    (void)hipMemsetAsync(S, 0, (size_t)G * DIM * 4, stream);
    (void)hipMemsetAsync(CNT, 0, (size_t)G * 4, stream);
    k_pool<<<(int)(((size_t)N * 32 + 255) / 256), 256, 0, stream>>>(cur, batch, S,
                                                                    CNT, N);
    k_div<<<(G * DIM + 255) / 256, 256, 0, stream>>>(S, CNT, (float*)d_out,
                                                     G * DIM);
}
